// ALEGrid_51685636440548
// MI455X (gfx1250) — matrix operand order repaired and checked by reading, not compiled, not run on hardware
//
#include <hip/hip_runtime.h>
#include <hip/hip_bf16.h>
#include <stdint.h>

#define L_   2
#define B_   4
#define N_   2048
#define G_   4096
#define HID_ 64
#define K_   5

typedef __attribute__((ext_vector_type(16))) __bf16 v16bf;
typedef __attribute__((ext_vector_type(8)))  float  v8f;

union FragBF { v16bf v; uint4 q[2]; };

__device__ __forceinline__ unsigned short f2bf(float f) {
  unsigned u = __float_as_uint(f);
  u += 0x7FFFu + ((u >> 16) & 1u);          // round-to-nearest-even
  return (unsigned short)(u >> 16);
}
__device__ __forceinline__ float gcoord(int v) {
  return -3.5f + (7.0f * (float)v) / 63.0f; // linspace(-3.5, 3.5, 64)
}

// Async DMA of one 32-byte chunk (two B128 beats) global -> LDS.
// ISA: INST_OFFSET is added to BOTH the LDS address (VDST vgpr) and the
// global address (VADDR vgpr pair), so one pair of operands covers both beats.
__device__ __forceinline__ void async_copy32(unsigned lds_off, const void* gptr) {
  uint64_t ga = (uint64_t)gptr;
  asm volatile(
      "global_load_async_to_lds_b128 %0, %1, off\n\t"
      "global_load_async_to_lds_b128 %0, %1, off offset:16"
      :: "v"(lds_off), "v"(ga) : "memory");
}
__device__ __forceinline__ void wait_async0() {
  asm volatile("s_wait_asynccnt 0" ::: "memory");
}

// ---------------- one-time bf16 convert of w_W ----------------
__global__ __launch_bounds__(256) void k_cvt(const float* __restrict__ in,
                                             unsigned short* __restrict__ out) {
  size_t i = (size_t)blockIdx.x * 256 + threadIdx.x;
  out[i] = f2bf(in[i]);
}

// ---------------- fused -cdist^2 + LayerNorm(G) -> bf16 row ----------------
__global__ __launch_bounds__(256) void k_ln_rows(const float* __restrict__ u, int l,
                                                 const float* __restrict__ gamma,
                                                 const float* __restrict__ beta,
                                                 unsigned short* __restrict__ lnbuf) {
  __shared__ float r1[256], r2[256];
  const int row = blockIdx.x;             // b*N + n
  const int b = row >> 11, n = row & (N_ - 1);
  const int tid = threadIdx.x;
  const float ux = u[(((size_t)l * B_ + b) * N_ + n) * 2 + 0];
  const float uy = u[(((size_t)l * B_ + b) * N_ + n) * 2 + 1];
  const float squ = ux * ux + uy * uy;
  float x[16], s1 = 0.f, s2 = 0.f;
#pragma unroll
  for (int i = 0; i < 16; ++i) {
    int g = i * 256 + tid;
    float gx = gcoord(g >> 6), gy = gcoord(g & 63);
    float v = -(squ + gx * gx + gy * gy - 2.0f * (ux * gx + uy * gy));
    x[i] = v; s1 += v; s2 += v * v;
  }
  r1[tid] = s1; r2[tid] = s2; __syncthreads();
  for (int s = 128; s > 0; s >>= 1) {
    if (tid < s) { r1[tid] += r1[tid + s]; r2[tid] += r2[tid + s]; }
    __syncthreads();
  }
  const float mu = r1[0] * (1.0f / G_);
  const float var = r2[0] * (1.0f / G_) - mu * mu;
  const float rstd = rsqrtf(var + 1e-5f);
  unsigned short* out = lnbuf + (size_t)row * G_;
#pragma unroll
  for (int i = 0; i < 16; ++i) {
    int g = i * 256 + tid;
    out[g] = f2bf((x[i] - mu) * rstd * gamma[g] + beta[g]);
  }
}

// ---------------- 128x128 tiled bf16 WMMA GEMM: S = A @ W^T + b ----------------
// Double-buffered LDS, staged with GLOBAL_LOAD_ASYNC_TO_LDS_B128 (ASYNCcnt).
// A: [M,4096] bf16 row-major.  Bt: [4096,4096] bf16 row-major (row j = w_W[j,:],
// i.e. exactly BT[n][k]).  S: [M,4096] f32.
__global__ __launch_bounds__(256) void k_gemm_bf16(const unsigned short* __restrict__ A,
                                                   const unsigned short* __restrict__ Bt,
                                                   const float* __restrict__ bias,
                                                   float* __restrict__ S) {
  __shared__ unsigned short lA[2][128 * 40];   // row stride 40 ushort (80B, 16B aligned)
  __shared__ unsigned short lB[2][128 * 40];
  const int row0 = blockIdx.y * 128, col0 = blockIdx.x * 128;
  const int tid = threadIdx.x, wave = tid >> 5, lane = tid & 31;
  const int wm = (wave >> 1) * 32, wn = (wave & 1) * 64;
  const int ml = lane & 15;
  const int kb  = (lane >> 4) * 8;    // A fragment k-base (ISA 16-bit A 16x32 layout)
  const int kb2 = (lane >> 4) * 16;   // unused: B fragments are loaded with kb, same k order as A
  const int r = tid >> 1, h = tid & 1;   // staging: 128 rows x 2 half-rows (32B each)

  const unsigned ldsA[2] = { (unsigned)(uintptr_t)(lA[0] + r * 40 + h * 16),
                             (unsigned)(uintptr_t)(lA[1] + r * 40 + h * 16) };
  const unsigned ldsB[2] = { (unsigned)(uintptr_t)(lB[0] + r * 40 + h * 16),
                             (unsigned)(uintptr_t)(lB[1] + r * 40 + h * 16) };
  const unsigned short* gA = A  + (size_t)(row0 + r) * 4096 + h * 16;
  const unsigned short* gB = Bt + (size_t)(col0 + r) * 4096 + h * 16;

  v8f zero;
#pragma unroll
  for (int i = 0; i < 8; ++i) zero[i] = 0.0f;
  v8f acc[2][4];
#pragma unroll
  for (int mi = 0; mi < 2; ++mi)
#pragma unroll
    for (int ni = 0; ni < 4; ++ni) acc[mi][ni] = zero;

  // prologue: stage K-step 0 into buffer 0
  async_copy32(ldsA[0], gA);
  async_copy32(ldsB[0], gB);

  for (int k0 = 0; k0 < 4096; k0 += 32) {
    const int cur = (k0 >> 5) & 1, nxt = cur ^ 1;
    wait_async0();        // my async writes of buf[cur] landed in LDS
    __syncthreads();      // all waves: buf[cur] ready, buf[nxt] fully consumed
    if (k0 + 32 < 4096) { // overlap next tile's DMA with this tile's WMMAs
      async_copy32(ldsA[nxt], gA + (k0 + 32));
      async_copy32(ldsB[nxt], gB + (k0 + 32));
    }
    FragBF fa[2], fb[4];
#pragma unroll
    for (int mi = 0; mi < 2; ++mi) {
      const int base = (wm + mi * 16 + ml) * 40 + kb;
      fa[mi].q[0] = *(const uint4*)(lA[cur] + base);        // K = kb..kb+7
      fa[mi].q[1] = *(const uint4*)(lA[cur] + base + 16);   // K = 16+kb..16+kb+7
    }
#pragma unroll
    for (int ni = 0; ni < 4; ++ni) {
      const int base = (wn + ni * 16 + ml) * 40 + kb;
      fb[ni].q[0] = *(const uint4*)(lB[cur] + base);        // K = kb..kb+7
      fb[ni].q[1] = *(const uint4*)(lB[cur] + base + 16);   // K = 16+kb..16+kb+7
    }
#pragma unroll
    for (int mi = 0; mi < 2; ++mi)
#pragma unroll
      for (int ni = 0; ni < 4; ++ni)
        acc[mi][ni] = __builtin_amdgcn_wmma_f32_16x16x32_bf16(
            false, fa[mi].v, false, fb[ni].v, (short)0, acc[mi][ni], false, false);
  }
  const int mh = (lane >> 4) * 8;   // C/D layout: lanes 16-31 hold M = v+8
#pragma unroll
  for (int ni = 0; ni < 4; ++ni) {
    const int c = col0 + wn + ni * 16 + ml;
    const float bv = bias[c];
#pragma unroll
    for (int mi = 0; mi < 2; ++mi)
#pragma unroll
      for (int v = 0; v < 8; ++v) {
        const int rr = row0 + wm + mi * 16 + mh + v;
        S[(size_t)rr * 4096 + c] = acc[mi][ni][v] + bv;
      }
  }
}

// ---------------- fused softmax over N + einsum('bng,bnd->bgd') - grid ----------------
__global__ __launch_bounds__(256) void k_softmax_off(const float* __restrict__ S,
                                                     const float* __restrict__ u, int l,
                                                     float* __restrict__ offs) {
  const int b = blockIdx.y;
  const int g = blockIdx.x * 256 + threadIdx.x;
  const float* col = S + (size_t)b * N_ * G_ + g;
  const float* ub = u + (((size_t)l * B_ + b) * N_) * 2;
  float m = -__builtin_inff(), denom = 0.f;
  for (int n = 0; n < N_; ++n) {
    float v = col[(size_t)n * G_];
    if (v > m) { denom = denom * expf(m - v) + 1.0f; m = v; }
    else       { denom += expf(v - m); }
  }
  float ax = 0.f, ay = 0.f;
  for (int n = 0; n < N_; ++n) {
    float e = expf(col[(size_t)n * G_] - m);
    ax += e * ub[n * 2 + 0];
    ay += e * ub[n * 2 + 1];
  }
  const float inv = 1.0f / denom;
  const float ox = ax * inv - gcoord(g >> 6);
  const float oy = ay * inv - gcoord(g & 63);
  const size_t o = ((size_t)b * G_ + g) * 2;
  if (l == 0) { offs[o] = ox;  offs[o + 1] = oy;  }
  else        { offs[o] += ox; offs[o + 1] += oy; }
}

// ---------------- LN(DIM=2) + fc, emit new_grid + bf16 copy + sqnorms ----------------
__global__ __launch_bounds__(256) void k_fc(const float* __restrict__ offs,
                                            const float* __restrict__ fcg,
                                            const float* __restrict__ fcbt,
                                            const float* __restrict__ fcW,
                                            const float* __restrict__ fcb,
                                            float* __restrict__ outg,
                                            unsigned short* __restrict__ xbf,
                                            float* __restrict__ sqn) {
  const int b = blockIdx.y;
  const int g = blockIdx.x * 256 + threadIdx.x;
  const size_t o = ((size_t)b * G_ + g) * 2;
  const float x0 = gcoord(g >> 6) + offs[o];
  const float x1 = gcoord(g & 63) + offs[o + 1];
  const float mu = 0.5f * (x0 + x1);
  const float d0 = x0 - mu, d1 = x1 - mu;
  const float rstd = rsqrtf(0.5f * (d0 * d0 + d1 * d1) + 1e-5f);
  const float y0 = d0 * rstd * fcg[0] + fcbt[0];
  const float y1 = d1 * rstd * fcg[1] + fcbt[1];
  float* op = outg + ((size_t)b * G_ + g) * HID_;
  unsigned short* xp = xbf + ((size_t)b * G_ + g) * HID_;
  float sq = 0.f;
#pragma unroll
  for (int h = 0; h < HID_; ++h) {
    float v = fcW[h * 2] * y0 + fcW[h * 2 + 1] * y1 + fcb[h];
    op[h] = v; xp[h] = f2bf(v); sq += v * v;
  }
  sqn[(size_t)b * G_ + g] = sq;
}

// ---------------- per-batch pairwise distance via bf16 WMMA gram ----------------
__global__ __launch_bounds__(256) void k_dist(const unsigned short* __restrict__ X,
                                              const float* __restrict__ sq,
                                              float* __restrict__ D) {
  __shared__ unsigned short lA[128 * 40];
  __shared__ unsigned short lB[128 * 40];
  const int row0 = blockIdx.y * 128, col0 = blockIdx.x * 128;
  const int tid = threadIdx.x, wave = tid >> 5, lane = tid & 31;
  const int wm = (wave >> 1) * 32, wn = (wave & 1) * 64;
  const int ml = lane & 15;
  const int kb = (lane >> 4) * 8, kb2 = (lane >> 4) * 16;
  v8f zero;
#pragma unroll
  for (int i = 0; i < 8; ++i) zero[i] = 0.0f;
  v8f acc[2][4];
#pragma unroll
  for (int mi = 0; mi < 2; ++mi)
#pragma unroll
    for (int ni = 0; ni < 4; ++ni) acc[mi][ni] = zero;

  for (int k0 = 0; k0 < HID_; k0 += 32) {
    __syncthreads();
    {
      const int r = tid >> 1, h = tid & 1;
      const uint4* ga = (const uint4*)(X + (size_t)(row0 + r) * HID_ + k0 + h * 16);
      uint4* la = (uint4*)(lA + r * 40 + h * 16);
      la[0] = ga[0]; la[1] = ga[1];
      const uint4* gb = (const uint4*)(X + (size_t)(col0 + r) * HID_ + k0 + h * 16);
      uint4* lb = (uint4*)(lB + r * 40 + h * 16);
      lb[0] = gb[0]; lb[1] = gb[1];
    }
    __syncthreads();
    FragBF fa[2], fb[4];
#pragma unroll
    for (int mi = 0; mi < 2; ++mi) {
      const int base = (wm + mi * 16 + ml) * 40 + kb;
      fa[mi].q[0] = *(const uint4*)(lA + base);
      fa[mi].q[1] = *(const uint4*)(lA + base + 16);
    }
#pragma unroll
    for (int ni = 0; ni < 4; ++ni) {
      const int base = (wn + ni * 16 + ml) * 40 + kb;
      fb[ni].q[0] = *(const uint4*)(lB + base);
      fb[ni].q[1] = *(const uint4*)(lB + base + 16);
    }
#pragma unroll
    for (int mi = 0; mi < 2; ++mi)
#pragma unroll
      for (int ni = 0; ni < 4; ++ni)
        acc[mi][ni] = __builtin_amdgcn_wmma_f32_16x16x32_bf16(
            false, fa[mi].v, false, fb[ni].v, (short)0, acc[mi][ni], false, false);
  }
  const int mh = (lane >> 4) * 8;
#pragma unroll
  for (int ni = 0; ni < 4; ++ni) {
    const int c = col0 + wn + ni * 16 + ml;
    const float sc = sq[c];
#pragma unroll
    for (int mi = 0; mi < 2; ++mi)
#pragma unroll
      for (int v = 0; v < 8; ++v) {
        const int rr = row0 + wm + mi * 16 + mh + v;
        float d = sq[rr] + sc - 2.0f * acc[mi][ni][v];
        if (rr == c) d = __builtin_inff();
        D[(size_t)rr * 4096 + c] = d;
      }
  }
}

// ---------------- per-row iterative top-K(=5) argmin, emit edge_index ----------------
__global__ __launch_bounds__(256) void k_topk(const float* __restrict__ D, int b,
                                              int* __restrict__ eout) {
  __shared__ float row[G_];
  __shared__ float rv[256];
  __shared__ int   ri[256];
  __shared__ int   sel[K_];
  const int i = blockIdx.x, tid = threadIdx.x;
#pragma unroll
  for (int t = 0; t < 16; ++t) row[t * 256 + tid] = D[(size_t)i * G_ + t * 256 + tid];
  __syncthreads();
  for (int kk = 0; kk < K_; ++kk) {
    float best = __builtin_inff(); int bi = G_;
#pragma unroll
    for (int t = 0; t < 16; ++t) {
      int j = t * 256 + tid;
      float v = row[j];
      if (v < best) { best = v; bi = j; }   // j increasing -> keeps lowest index on ties
    }
    rv[tid] = best; ri[tid] = bi; __syncthreads();
    for (int s = 128; s > 0; s >>= 1) {
      if (tid < s) {
        float v2 = rv[tid + s]; int i2 = ri[tid + s];
        if (v2 < rv[tid] || (v2 == rv[tid] && i2 < ri[tid])) { rv[tid] = v2; ri[tid] = i2; }
      }
      __syncthreads();
    }
    if (tid == 0) { sel[kk] = ri[0]; row[ri[0]] = __builtin_inff(); }
    __syncthreads();
  }
  if (tid < K_) {
    const int BGK = B_ * G_ * K_;
    const int pos = (b * G_ + i) * K_ + tid;
    eout[pos]       = sel[tid] + b * G_;   // src (neighbor)
    eout[BGK + pos] = i + b * G_;          // dst
  }
}

extern "C" void kernel_launch(void* const* d_in, const int* in_sizes, int n_in,
                              void* d_out, int out_size, void* d_ws, size_t ws_size,
                              hipStream_t stream) {
  (void)in_sizes; (void)n_in; (void)out_size; (void)ws_size;
  const float* u        = (const float*)d_in[0];
  const float* w_gamma  = (const float*)d_in[1];
  const float* w_beta   = (const float*)d_in[2];
  const float* w_W      = (const float*)d_in[3];
  const float* w_b      = (const float*)d_in[4];
  const float* fc_gamma = (const float*)d_in[5];
  const float* fc_beta  = (const float*)d_in[6];
  const float* fc_W     = (const float*)d_in[7];
  const float* fc_b     = (const float*)d_in[8];

  char* w = (char*)d_ws;
  unsigned short* lnbuf = (unsigned short*)w; w += (size_t)B_ * N_ * G_ * 2;   //  67 MB
  unsigned short* wWb   = (unsigned short*)w; w += (size_t)G_ * G_ * 2;        //  33 MB
  float* sbuf           = (float*)w;          w += (size_t)B_ * N_ * G_ * 4;   // 134 MB
  float* offs           = (float*)w;          w += (size_t)B_ * G_ * 2 * 4;
  unsigned short* xbf   = (unsigned short*)w; w += (size_t)B_ * G_ * HID_ * 2;
  float* sqn            = (float*)w;          w += (size_t)B_ * G_ * 4;
  float* dist           = (float*)w;          w += (size_t)G_ * G_ * 4;        //  67 MB (reused per batch)

  float* outg = (float*)d_out;
  int* eout = (int*)((float*)d_out + (size_t)B_ * G_ * HID_);

  k_cvt<<<(G_ * G_) / 256, 256, 0, stream>>>(w_W, wWb);

  for (int l = 0; l < L_; ++l) {
    k_ln_rows<<<B_ * N_, 256, 0, stream>>>(u, l, w_gamma, w_beta, lnbuf);
    k_gemm_bf16<<<dim3(G_ / 128, (B_ * N_) / 128), 256, 0, stream>>>(lnbuf, wWb, w_b, sbuf);
    k_softmax_off<<<dim3(G_ / 256, B_), 256, 0, stream>>>(sbuf, u, l, offs);
  }
  k_fc<<<dim3(G_ / 256, B_), 256, 0, stream>>>(offs, fc_gamma, fc_beta, fc_W, fc_b,
                                               outg, xbf, sqn);
  for (int b = 0; b < B_; ++b) {
    k_dist<<<dim3(G_ / 128, G_ / 128), 256, 0, stream>>>(
        xbf + (size_t)b * G_ * HID_, sqn + (size_t)b * G_, dist);
    k_topk<<<G_, 256, 0, stream>>>(dist, b, eout);
  }
}
